// RNN_2723009266518
// MI455X (gfx1250) — matrix operand order repaired and checked by reading, not compiled, not run on hardware
//
#include <hip/hip_runtime.h>
#include <hip/hip_bf16.h>
#include <math.h>

typedef __bf16 bf16_t;
typedef __attribute__((ext_vector_type(16))) __bf16 v16bf;
typedef __attribute__((ext_vector_type(8)))  float   v8f;

#define B_SZ   256
#define T_SZ   100
#define IN_SZ  2
#define H_SZ   4096
#define OUT_SZ 512
#define ALPHA  0.1f

#define LDS_PAD 16
#define LDS_W   (64 + LDS_PAD)   // 80 elems = 160B row stride (16B aligned)

union FragU { v16bf v; uint4 q[2]; };

// ---------------------------------------------------------------------------
// f32 -> bf16 conversion (one-time for W_rec / W_out)
// ---------------------------------------------------------------------------
__global__ void cvt_f32_bf16(const float* __restrict__ src,
                             bf16_t* __restrict__ dst, int n) {
  int i = blockIdx.x * blockDim.x + threadIdx.x;
  int stride = gridDim.x * blockDim.x;
  for (; i < n; i += stride) dst[i] = (bf16_t)src[i];
}

// ---------------------------------------------------------------------------
// h0 = place_cells_0 @ W_h_init.T   (256x512)@(512x4096) -> (256x4096)
// ---------------------------------------------------------------------------
__global__ void init_h_kernel(const float* __restrict__ pc,
                              const float* __restrict__ Whi,
                              float* __restrict__ h_f32,
                              bf16_t* __restrict__ h_bf16) {
  int idx = blockIdx.x * blockDim.x + threadIdx.x;   // over B*H
  int b = idx >> 12;
  int j = idx & (H_SZ - 1);
  const float* pcrow = pc  + (size_t)b * OUT_SZ;
  const float* wrow  = Whi + (size_t)j * OUT_SZ;
  float acc = 0.0f;
#pragma unroll 8
  for (int o = 0; o < OUT_SZ; ++o) acc += pcrow[o] * wrow[o];
  h_f32[idx]  = acc;
  h_bf16[idx] = (bf16_t)acc;
}

// ---------------------------------------------------------------------------
// Async stage of one 64-wide K chunk of A(64 rows) and B(64 rows) into LDS.
// 128 threads; each issues 8 global_load_async_to_lds_b128 (ASYNCcnt += 8/wave)
// ---------------------------------------------------------------------------
__device__ __forceinline__ void stage_async(
    const bf16_t* __restrict__ A, const bf16_t* __restrict__ Bm,
    int tile_m, int tile_n, int kc,
    bf16_t (&As)[64][LDS_W], bf16_t (&Bs)[64][LDS_W], int tid)
{
#pragma unroll
  for (int p = 0; p < 4; ++p) {
    const int slot = tid + p * 128;        // 512 slots of 16B
    const int row  = slot >> 3;
    const int col  = (slot & 7) * 8;
    const bf16_t* g = &A[(size_t)(tile_m + row) * H_SZ + kc + col];
    const unsigned l = (unsigned)(uintptr_t)&As[row][col];
    asm volatile("global_load_async_to_lds_b128 %0, %1, off"
                 :: "v"(l), "v"(g) : "memory");
  }
#pragma unroll
  for (int p = 0; p < 4; ++p) {
    const int slot = tid + p * 128;
    const int row  = slot >> 3;
    const int col  = (slot & 7) * 8;
    const bf16_t* g = &Bm[(size_t)(tile_n + row) * H_SZ + kc + col];
    const unsigned l = (unsigned)(uintptr_t)&Bs[row][col];
    asm volatile("global_load_async_to_lds_b128 %0, %1, off"
                 :: "v"(l), "v"(g) : "memory");
  }
}

// ---------------------------------------------------------------------------
// Per-wave 32x32 MMA on one staged 64-K chunk (2 k-steps of 32).
// Fragment layouts per CDNA5 ISA 7.12.2:
//   A lane(m=l%16): K runs [k0..k0+7],[k0+16..k0+23], k0=(l/16)*8
//   B lane(n=l%16): K runs [kb..kb+7],[kb+16..kb+23], kb=(l/16)*8
// ---------------------------------------------------------------------------
__device__ __forceinline__ void compute_chunk(
    const bf16_t (&As)[64][LDS_W], const bf16_t (&Bs)[64][LDS_W],
    int wm, int wn, int lane,
    v8f& c00, v8f& c01, v8f& c10, v8f& c11)
{
  const int mA   = lane & 15;
  const int a_k0 = (lane >> 4) * 8;
  const int b_k0 = (lane >> 4) * 8;
#pragma unroll
  for (int ks = 0; ks < 2; ++ks) {
    const int kk = ks * 32;
    FragU a0, a1, b0, b1;
    a0.q[0] = *(const uint4*)&As[wm + mA][kk + a_k0];
    a0.q[1] = *(const uint4*)&As[wm + mA][kk + a_k0 + 16];
    a1.q[0] = *(const uint4*)&As[wm + 16 + mA][kk + a_k0];
    a1.q[1] = *(const uint4*)&As[wm + 16 + mA][kk + a_k0 + 16];
    b0.q[0] = *(const uint4*)&Bs[wn + mA][kk + b_k0];
    b0.q[1] = *(const uint4*)&Bs[wn + mA][kk + b_k0 + 16];
    b1.q[0] = *(const uint4*)&Bs[wn + 16 + mA][kk + b_k0];
    b1.q[1] = *(const uint4*)&Bs[wn + 16 + mA][kk + b_k0 + 16];

    c00 = __builtin_amdgcn_wmma_f32_16x16x32_bf16(false, a0.v, false, b0.v,
                                                  (short)0, c00, false, false);
    c01 = __builtin_amdgcn_wmma_f32_16x16x32_bf16(false, a0.v, false, b1.v,
                                                  (short)0, c01, false, false);
    c10 = __builtin_amdgcn_wmma_f32_16x16x32_bf16(false, a1.v, false, b0.v,
                                                  (short)0, c10, false, false);
    c11 = __builtin_amdgcn_wmma_f32_16x16x32_bf16(false, a1.v, false, b1.v,
                                                  (short)0, c11, false, false);
  }
}

// ---------------------------------------------------------------------------
// One recurrence step: pre = x_t + h @ W_rec.T + (b_in + b_rec)
//                      h'  = (1-a)*h + a*tanh(pre)
// 64x64 C tile per 128-thread block (4 waves, 2x2, 32x32 each).
// Double-buffered async LDS pipeline over K=4096 in 64-chunks.
// ---------------------------------------------------------------------------
__global__ __launch_bounds__(128)
void rnn_step_kernel(const bf16_t* __restrict__ hA, const float* __restrict__ hF,
                     const bf16_t* __restrict__ Wr,
                     const float* __restrict__ inputs, const float* __restrict__ W_in,
                     const float* __restrict__ b_in, const float* __restrict__ b_rec,
                     int t,
                     float* __restrict__ hidden,
                     float* __restrict__ hF_out, bf16_t* __restrict__ hA_out)
{
  __shared__ bf16_t As[2][64][LDS_W];
  __shared__ bf16_t Bs[2][64][LDS_W];

  const int tid    = threadIdx.x;
  const int lane   = tid & 31;
  const int wave   = tid >> 5;
  const int wm     = (wave & 1) * 32;
  const int wn     = (wave >> 1) * 32;
  const int tile_m = blockIdx.y * 64;
  const int tile_n = blockIdx.x * 64;

  v8f c00 = {}, c01 = {}, c10 = {}, c11 = {};

  stage_async(hA, Wr, tile_m, tile_n, 0, As[0], Bs[0], tid);

  const int NCH = H_SZ / 64;
  for (int i = 0; i < NCH; ++i) {
    if (i + 1 < NCH) {   // uniform branch
      stage_async(hA, Wr, tile_m, tile_n, (i + 1) * 64,
                  As[(i + 1) & 1], Bs[(i + 1) & 1], tid);
      asm volatile("s_wait_asynccnt 0x8" ::: "memory");  // chunk i landed
    } else {
      asm volatile("s_wait_asynccnt 0x0" ::: "memory");
    }
    __syncthreads();                         // chunk i visible to all waves
    compute_chunk(As[i & 1], Bs[i & 1], wm, wn, lane, c00, c01, c10, c11);
    __syncthreads();                         // buffer free before re-stage
  }

  // ---- fused epilogue: input projection (IN=2), biases, tanh, leaky update
  const int n     = lane & 15;
  const int rhalf = (lane >> 4) * 8;     // C layout: lanes>=16 hold M=8..15
#pragma unroll
  for (int ni = 0; ni < 2; ++ni) {
    const int   jg   = tile_n + wn + ni * 16 + n;
    const float xw0  = W_in[jg * IN_SZ + 0];
    const float xw1  = W_in[jg * IN_SZ + 1];
    const float bias = b_in[jg] + b_rec[jg];
#pragma unroll
    for (int mi = 0; mi < 2; ++mi) {
      v8f acc = (mi == 0) ? (ni == 0 ? c00 : c01) : (ni == 0 ? c10 : c11);
      const int mbase = tile_m + wm + mi * 16 + rhalf;
#pragma unroll
      for (int r = 0; r < 8; ++r) {
        const int bg = mbase + r;
        const float x0 = inputs[(size_t)bg * (T_SZ * IN_SZ) + t * IN_SZ + 0];
        const float x1 = inputs[(size_t)bg * (T_SZ * IN_SZ) + t * IN_SZ + 1];
        const float pre  = acc[r] + x0 * xw0 + x1 * xw1 + bias;
        const float hnew = (1.0f - ALPHA) * hF[(size_t)bg * H_SZ + jg]
                         + ALPHA * tanhf(pre);
        hF_out[(size_t)bg * H_SZ + jg] = hnew;
        hA_out[(size_t)bg * H_SZ + jg] = (bf16_t)hnew;
        hidden[(size_t)bg * (T_SZ * H_SZ) + (size_t)t * H_SZ + jg] = hnew;
      }
    }
  }
}

// ---------------------------------------------------------------------------
// outputs = hidden @ W_out.T as one GEMM: (25600 x 4096) @ (4096 x 512).
// hidden(B,T,H) is row-major (B*T, H); outputs(B,T,OUT) is (B*T, OUT).
// A is converted f32->bf16 while staging into LDS.
// ---------------------------------------------------------------------------
__global__ __launch_bounds__(128)
void out_gemm_kernel(const float* __restrict__ hidden,
                     const bf16_t* __restrict__ Wo,
                     float* __restrict__ outp)
{
  __shared__ bf16_t As[64][LDS_W];
  __shared__ bf16_t Bs[64][LDS_W];

  const int tid    = threadIdx.x;
  const int lane   = tid & 31;
  const int wave   = tid >> 5;
  const int wm     = (wave & 1) * 32;
  const int wn     = (wave >> 1) * 32;
  const int tile_m = blockIdx.y * 64;   // over B*T rows
  const int tile_n = blockIdx.x * 64;   // over OUT cols

  v8f c00 = {}, c01 = {}, c10 = {}, c11 = {};

  for (int kc = 0; kc < H_SZ; kc += 64) {
    __syncthreads();   // previous chunk's fragment reads done
    // A tile: 64 x 64 f32 -> bf16
#pragma unroll
    for (int p = 0; p < 8; ++p) {
      const int slot = tid + p * 128;       // 1024 slots of 4 f32
      const int row  = slot >> 4;
      const int col  = (slot & 15) * 4;
      float4 v = *(const float4*)&hidden[(size_t)(tile_m + row) * H_SZ + kc + col];
      bf16_t tmp[4] = {(bf16_t)v.x, (bf16_t)v.y, (bf16_t)v.z, (bf16_t)v.w};
      *(uint2*)&As[row][col] = *(const uint2*)tmp;
    }
    // B tile: 64 x 64 bf16
#pragma unroll
    for (int p = 0; p < 4; ++p) {
      const int slot = tid + p * 128;
      const int row  = slot >> 3;
      const int col  = (slot & 7) * 8;
      uint4 v = *(const uint4*)&Wo[(size_t)(tile_n + row) * H_SZ + kc + col];
      if (kc + 64 < H_SZ)
        __builtin_prefetch(&Wo[(size_t)(tile_n + row) * H_SZ + kc + 64 + col], 0, 1);
      *(uint4*)&Bs[row][col] = v;
    }
    __syncthreads();
    compute_chunk(As, Bs, wm, wn, lane, c00, c01, c10, c11);
  }

  const int n     = lane & 15;
  const int rhalf = (lane >> 4) * 8;
#pragma unroll
  for (int ni = 0; ni < 2; ++ni) {
    const int og = tile_n + wn + ni * 16 + n;
#pragma unroll
    for (int mi = 0; mi < 2; ++mi) {
      v8f acc = (mi == 0) ? (ni == 0 ? c00 : c01) : (ni == 0 ? c10 : c11);
      const int mbase = tile_m + wm + mi * 16 + rhalf;
#pragma unroll
      for (int r = 0; r < 8; ++r)
        outp[(size_t)(mbase + r) * OUT_SZ + og] = acc[r];
    }
  }
}

// ---------------------------------------------------------------------------
extern "C" void kernel_launch(void* const* d_in, const int* in_sizes, int n_in,
                              void* d_out, int out_size, void* d_ws, size_t ws_size,
                              hipStream_t stream)
{
  (void)in_sizes; (void)n_in; (void)out_size; (void)ws_size;

  const float* inputs = (const float*)d_in[0];
  const float* pc0    = (const float*)d_in[1];
  const float* W_in   = (const float*)d_in[2];
  const float* b_in   = (const float*)d_in[3];
  const float* W_rec  = (const float*)d_in[4];
  const float* b_rec  = (const float*)d_in[5];
  const float* W_out  = (const float*)d_in[6];
  const float* W_hi   = (const float*)d_in[7];

  float* hidden  = (float*)d_out;                          // (B,T,H)
  float* outputs = hidden + (size_t)B_SZ * T_SZ * H_SZ;    // (B,T,OUT)

  // workspace layout (~48 MB total)
  uint8_t* ws = (uint8_t*)d_ws;
  size_t off = 0;
  bf16_t* Wr_bf = (bf16_t*)(ws + off); off += (size_t)H_SZ * H_SZ * 2;    // 32 MB
  bf16_t* Wo_bf = (bf16_t*)(ws + off); off += (size_t)OUT_SZ * H_SZ * 2;  //  4 MB
  float*  hF0   = (float* )(ws + off); off += (size_t)B_SZ * H_SZ * 4;    //  4 MB
  float*  hF1   = (float* )(ws + off); off += (size_t)B_SZ * H_SZ * 4;    //  4 MB
  bf16_t* hB0   = (bf16_t*)(ws + off); off += (size_t)B_SZ * H_SZ * 2;    //  2 MB
  bf16_t* hB1   = (bf16_t*)(ws + off); off += (size_t)B_SZ * H_SZ * 2;    //  2 MB
  float*  hFb[2] = {hF0, hF1};
  bf16_t* hBb[2] = {hB0, hB1};

  cvt_f32_bf16<<<2048, 256, 0, stream>>>(W_rec, Wr_bf, H_SZ * H_SZ);
  cvt_f32_bf16<<<512,  256, 0, stream>>>(W_out, Wo_bf, OUT_SZ * H_SZ);
  init_h_kernel<<<(B_SZ * H_SZ) / 256, 256, 0, stream>>>(pc0, W_hi, hF0, hB0);

  for (int t = 0; t < T_SZ; ++t) {
    const int cur = t & 1;
    const int nxt = (t + 1) & 1;
    rnn_step_kernel<<<dim3(H_SZ / 64, B_SZ / 64), 128, 0, stream>>>(
        hBb[cur], hFb[cur], Wr_bf, inputs, W_in, b_in, b_rec, t,
        hidden, hFb[nxt], hBb[nxt]);
  }
  out_gemm_kernel<<<dim3(OUT_SZ / 64, (B_SZ * T_SZ) / 64), 128, 0, stream>>>(
      hidden, Wo_bf, outputs);
}
